// GCNNetV2_73564199846476
// MI455X (gfx1250) — matrix operand order repaired and checked by reading, not compiled, not run on hardware
//
#include <hip/hip_runtime.h>
#include <hip/hip_bf16.h>
#include <stdint.h>

#define NN 100000
#define NE 400000
#define NG 4000
#define NF 39
#define BN_EPS 1e-5f

typedef __attribute__((ext_vector_type(16))) __bf16 v16bf;
typedef __attribute__((ext_vector_type(8)))  float  v8f;

union Frag { v16bf v; uint4 u[2]; };

__device__ __forceinline__ unsigned short f2bf(float f) {
  unsigned u = __float_as_uint(f);
  unsigned r = u + 0x7FFFu + ((u >> 16) & 1u);   // RNE to bf16
  return (unsigned short)(r >> 16);
}
__device__ __forceinline__ float bf2f(unsigned short s) {
  return __uint_as_float(((unsigned)s) << 16);
}
// order-preserving float<->uint for atomicMax on signed floats
__device__ __forceinline__ unsigned encf(float f) {
  unsigned u = __float_as_uint(f);
  return (u & 0x80000000u) ? ~u : (u | 0x80000000u);
}
__device__ __forceinline__ float decf(unsigned u) {
  unsigned v = (u & 0x80000000u) ? (u & 0x7FFFFFFFu) : ~u;
  return __uint_as_float(v);
}

// ---------------- utility ----------------
__global__ void k_fill(unsigned* p, long long n, unsigned v) {
  long long i = (long long)blockIdx.x * blockDim.x + threadIdx.x;
  if (i < n) p[i] = v;
}

// ---------------- graph normalization ----------------
__global__ void k_deg(const int* col, unsigned* deg, int E) {
  int i = blockIdx.x * blockDim.x + threadIdx.x;
  if (i < E) atomicAdd(&deg[col[i]], 1u);
}
__global__ void k_dis(const unsigned* deg, float* dis, int n) {
  int i = blockIdx.x * blockDim.x + threadIdx.x;
  if (i < n) dis[i] = rsqrtf((float)deg[i]);   // deg >= 1 (self loop)
}
__global__ void k_enorm(const int* row, const int* col, const float* dis, float* enorm, int E) {
  int i = blockIdx.x * blockDim.x + threadIdx.x;
  if (i < E) enorm[i] = dis[row[i]] * dis[col[i]];
}

// ---------------- bf16 split conversion ----------------
// activations: [n, din] fp32 -> [n, dinp] bf16 hi/lo (K zero-padded)
__global__ void k_split(const float* src, int n, int din, int dinp,
                        unsigned short* hi, unsigned short* lo) {
  long long idx = (long long)blockIdx.x * blockDim.x + threadIdx.x;
  long long tot = (long long)n * dinp;
  if (idx >= tot) return;
  int k = (int)(idx % dinp);
  long long i = idx / dinp;
  float v = (k < din) ? src[i * din + k] : 0.f;
  unsigned short h = f2bf(v);
  hi[idx] = h;
  lo[idx] = f2bf(v - bf2f(h));
}
// weights: W[din, dout] fp32 -> Wt[dout, dinp] bf16 hi/lo (transposed, K-padded)
__global__ void k_wt(const float* W, int din, int dinp, int dout,
                     unsigned short* hiT, unsigned short* loT) {
  long long idx = (long long)blockIdx.x * blockDim.x + threadIdx.x;
  long long tot = (long long)dout * dinp;
  if (idx >= tot) return;
  int k = (int)(idx % dinp);
  int o = (int)(idx / dinp);
  float v = (k < din) ? W[(size_t)k * dout + o] : 0.f;
  unsigned short h = f2bf(v);
  hiT[idx] = h;
  loT[idx] = f2bf(v - bf2f(h));
}

// ---------------- WMMA bf16x3 GEMM: H = X @ W (raw, bias added later) ----------------
__global__ __launch_bounds__(256) void k_gemm(
    const unsigned short* __restrict__ Xhi, const unsigned short* __restrict__ Xlo,
    const unsigned short* __restrict__ WhiT, const unsigned short* __restrict__ WloT,
    float* __restrict__ H, int nRows, int dinp, int dout) {
  int nRowT = nRows >> 4;
  int nColT = dout >> 4;
  int wave = blockIdx.x * (blockDim.x >> 5) + (threadIdx.x >> 5);
  if (wave >= nRowT * nColT) return;              // wave-uniform
  int lane = threadIdx.x & 31;
  int rowT = wave % nRowT;
  int colT = wave / nRowT;
  int m = lane & 15;          // A row / B col / D col (per lane)
  int half = lane >> 4;

  // A fragment: elems 0..7 -> K = k0 + 8*half + j ; elems 8..15 -> K = k0 + 16 + 8*half + j
  const unsigned short* aHiP = Xhi + (size_t)(rowT * 16 + m) * dinp + 8 * half;
  const unsigned short* aLoP = Xlo + (size_t)(rowT * 16 + m) * dinp + 8 * half;
  // B fragment from Wt[dout, dinp]: elems 0..7 -> K = k0 + 8*half + j ; elems 8..15 -> K = k0 + 16 + 8*half + j
  const unsigned short* bHiP = WhiT + (size_t)(colT * 16 + m) * dinp + 8 * half;
  const unsigned short* bLoP = WloT + (size_t)(colT * 16 + m) * dinp + 8 * half;

  v8f acc = {0.f, 0.f, 0.f, 0.f, 0.f, 0.f, 0.f, 0.f};
  for (int k0 = 0; k0 < dinp; k0 += 32) {
    Frag ah, al, bh, blo;
    ah.u[0] = *(const uint4*)(aHiP + k0);
    ah.u[1] = *(const uint4*)(aHiP + k0 + 16);
    al.u[0] = *(const uint4*)(aLoP + k0);
    al.u[1] = *(const uint4*)(aLoP + k0 + 16);
    bh.u[0] = *(const uint4*)(bHiP + k0);
    bh.u[1] = *(const uint4*)(bHiP + k0 + 16);
    blo.u[0] = *(const uint4*)(bLoP + k0);
    blo.u[1] = *(const uint4*)(bLoP + k0 + 16);
    acc = __builtin_amdgcn_wmma_f32_16x16x32_bf16(false, ah.v, false, bh.v,  (short)0, acc, false, false);
    acc = __builtin_amdgcn_wmma_f32_16x16x32_bf16(false, ah.v, false, blo.v, (short)0, acc, false, false);
    acc = __builtin_amdgcn_wmma_f32_16x16x32_bf16(false, al.v, false, bh.v,  (short)0, acc, false, false);
  }
  // D layout: VGPR r -> M = r + 8*half, N = lane&15
  float* outp = H + (size_t)(rowT * 16 + 8 * half) * dout + colT * 16 + m;
#pragma unroll
  for (int r = 0; r < 8; ++r) outp[(size_t)r * dout] = acc[r];
}

// ---------------- message passing ----------------
// self-loop init (non-atomic): O = dis^2 * H + bias
__global__ void k_selfinit(const float* H, const float* dis, const float* bias,
                           float* O, int n, int dout) {
  long long idx = (long long)blockIdx.x * blockDim.x + threadIdx.x;
  long long tot = (long long)n * dout;
  if (idx >= tot) return;
  int f = (int)(idx % dout);
  long long i = idx / dout;
  float d = dis[i];
  O[idx] = d * d * H[idx] + bias[f];
}
// edge scatter: O[col] += norm * H[row]
__global__ void k_scatter(const int* __restrict__ row, const int* __restrict__ col,
                          const float* __restrict__ enorm, const float* __restrict__ H,
                          float* __restrict__ O, int E, int dout) {
  long long idx = (long long)blockIdx.x * blockDim.x + threadIdx.x;
  int q = dout >> 2;
  long long tot = (long long)E * q;
  if (idx >= tot) return;
  int e = (int)(idx / q);
  int f4 = (int)(idx % q);
  int r = row[e], c = col[e];
  float w = enorm[e];
  const float4 hv = *(const float4*)(H + (size_t)r * dout + f4 * 4);
  float* op = O + (size_t)c * dout + f4 * 4;
  atomicAdd(op + 0, w * hv.x);
  atomicAdd(op + 1, w * hv.y);
  atomicAdd(op + 2, w * hv.z);
  atomicAdd(op + 3, w * hv.w);
}

// ---------------- BatchNorm(ReLU(x)) ----------------
__global__ __launch_bounds__(256) void k_bnstats(const float* O, int n, int dout,
                                                 float* s1g, float* s2g) {
  int nc = dout >> 8;   // columns per thread: 1,2,4 (dout in {256,512,1024})
  float a1[4] = {0.f, 0.f, 0.f, 0.f}, a2[4] = {0.f, 0.f, 0.f, 0.f};
  int r0 = blockIdx.x * 256;
  int r1 = (r0 + 256 < n) ? (r0 + 256) : n;
  for (int r = r0; r < r1; ++r) {
    const float* rp = O + (size_t)r * dout;
#pragma unroll
    for (int j = 0; j < 4; ++j) {
      if (j < nc) {
        float v = fmaxf(rp[threadIdx.x + (j << 8)], 0.f);
        a1[j] += v;
        a2[j] += v * v;
      }
    }
  }
#pragma unroll
  for (int j = 0; j < 4; ++j) {
    if (j < nc) {
      atomicAdd(&s1g[threadIdx.x + (j << 8)], a1[j]);
      atomicAdd(&s2g[threadIdx.x + (j << 8)], a2[j]);
    }
  }
}
__global__ void k_bnfinal(const float* s1, const float* s2, const float* g, const float* be,
                          float* scale, float* shift, int dout, int n) {
  int t = blockIdx.x * blockDim.x + threadIdx.x;
  if (t >= dout) return;
  float invn = 1.f / (float)n;
  float mean = s1[t] * invn;
  float var = s2[t] * invn - mean * mean;
  float sc = g[t] * rsqrtf(var + BN_EPS);
  scale[t] = sc;
  shift[t] = be[t] - mean * sc;
}
__global__ void k_bnapply(float* O, const float* scale, const float* shift, int n, int dout) {
  long long idx = (long long)blockIdx.x * blockDim.x + threadIdx.x;
  long long tot = (long long)n * dout;
  if (idx >= tot) return;
  int f = (int)(idx % dout);
  O[idx] = fmaxf(O[idx], 0.f) * scale[f] + shift[f];
}

// ---------------- pooling + head ----------------
// wave-per-output FC: out[g,o] = (relu?)(in[g,:] . W[:,o] + b[o])
__global__ void k_fcw(const float* __restrict__ in, const float* __restrict__ W,
                      const float* __restrict__ b, float* __restrict__ out,
                      int G, int din, int dout, int relu) {
  int wave = blockIdx.x * (blockDim.x >> 5) + (threadIdx.x >> 5);
  int lane = threadIdx.x & 31;
  if (wave >= G * dout) return;
  int g = wave / dout, o = wave % dout;
  float s = 0.f;
  for (int k = lane; k < din; k += 32) s += in[(size_t)g * din + k] * W[(size_t)k * dout + o];
  for (int off = 16; off; off >>= 1) s += __shfl_xor(s, off, 32);
  if (lane == 0) {
    s += b[o];
    if (relu) s = fmaxf(s, 0.f);
    out[(size_t)g * dout + o] = s;
  }
}
__global__ void k_gmax(const float* gate, const int* batch, unsigned* gmU, int n) {
  int i = blockIdx.x * blockDim.x + threadIdx.x;
  if (i < n) atomicMax(&gmU[batch[i]], encf(gate[i]));
}
__global__ void k_exp(const float* gate, const int* batch, const unsigned* gmU,
                      float* eex, float* denom, int n) {
  int i = blockIdx.x * blockDim.x + threadIdx.x;
  if (i >= n) return;
  int b = batch[i];
  float e = expf(gate[i] - decf(gmU[b]));
  eex[i] = e;
  atomicAdd(&denom[b], e);
}
__global__ void k_alpha(const float* eex, const float* denom, const int* batch,
                        float* alpha, int n) {
  int i = blockIdx.x * blockDim.x + threadIdx.x;
  if (i < n) alpha[i] = eex[i] / denom[batch[i]];
}
__global__ void k_pool(const float* __restrict__ X, const float* __restrict__ alpha,
                       const int* __restrict__ batch, float* __restrict__ pooled, int n) {
  long long idx = (long long)blockIdx.x * blockDim.x + threadIdx.x;
  long long tot = (long long)n * 256;          // 1024 feats / 4 per thread
  if (idx >= tot) return;
  int i = (int)(idx >> 8);
  int t = (int)(idx & 255);
  int b = batch[i];
  float a = alpha[i];
  const float4 xv = *(const float4*)(X + (size_t)i * 1024 + t * 4);
  float* pp = pooled + (size_t)b * 1024 + t * 4;
  atomicAdd(pp + 0, a * xv.x);
  atomicAdd(pp + 1, a * xv.y);
  atomicAdd(pp + 2, a * xv.z);
  atomicAdd(pp + 3, a * xv.w);
}

extern "C" void kernel_launch(void* const* d_in, const int* in_sizes, int n_in,
                              void* d_out, int out_size, void* d_ws, size_t ws_size,
                              hipStream_t stream) {
  (void)in_sizes; (void)n_in; (void)out_size; (void)ws_size;
  const float* x0 = (const float*)d_in[0];
  const int* ei = (const int*)d_in[1];
  const int* erow = ei;
  const int* ecol = ei + NE;
  const int* batch = (const int*)d_in[2];
  const float *Wl[5], *bl[5], *gl[5], *bel[5];
  for (int l = 0; l < 5; ++l) {
    Wl[l]  = (const float*)d_in[3 + 4 * l];
    bl[l]  = (const float*)d_in[4 + 4 * l];
    gl[l]  = (const float*)d_in[5 + 4 * l];
    bel[l] = (const float*)d_in[6 + 4 * l];
  }
  const float* gateW = (const float*)d_in[23];
  const float* gateB = (const float*)d_in[24];
  const float* fc2W = (const float*)d_in[25];
  const float* fc2b = (const float*)d_in[26];
  const float* fc3W = (const float*)d_in[27];
  const float* fc3b = (const float*)d_in[28];
  const float* fc4W = (const float*)d_in[29];
  const float* fc4b = (const float*)d_in[30];

  size_t off = 0;
  auto carve = [&](size_t bytes) -> void* {
    void* r = (char*)d_ws + off;
    off += (bytes + 255) & ~(size_t)255;
    return r;
  };
  unsigned* deg       = (unsigned*)carve((size_t)NN * 4);
  float* dis          = (float*)carve((size_t)NN * 4);
  float* enorm        = (float*)carve((size_t)NE * 4);
  float* bufX         = (float*)carve((size_t)NN * 1024 * 4);
  float* bufH         = (float*)carve((size_t)NN * 1024 * 4);
  unsigned short* Xhi = (unsigned short*)carve((size_t)NN * 1024 * 2);
  unsigned short* Xlo = (unsigned short*)carve((size_t)NN * 1024 * 2);
  unsigned short* WhiT = (unsigned short*)carve((size_t)1024 * 1024 * 2);
  unsigned short* WloT = (unsigned short*)carve((size_t)1024 * 1024 * 2);
  float* gsum   = (float*)carve(1024 * 4);
  float* gsqs   = (float*)carve(1024 * 4);
  float* bscale = (float*)carve(1024 * 4);
  float* bshift = (float*)carve(1024 * 4);
  float* gate   = (float*)carve((size_t)NN * 4);
  float* eex    = (float*)carve((size_t)NN * 4);
  unsigned* gmU = (unsigned*)carve((size_t)NG * 4);
  float* denom  = (float*)carve((size_t)NG * 4);
  float* pooled = (float*)carve((size_t)NG * 1024 * 4);
  float* h2     = (float*)carve((size_t)NG * 128 * 4);
  float* h3     = (float*)carve((size_t)NG * 16 * 4);

  const int T = 256;
  auto nb = [&](long long tot) { return (unsigned)((tot + T - 1) / T); };

  // graph norm
  k_fill<<<nb(NN), T, 0, stream>>>(deg, NN, 1u);                 // self-loop counts
  k_deg<<<nb(NE), T, 0, stream>>>(ecol, deg, NE);
  k_dis<<<nb(NN), T, 0, stream>>>(deg, dis, NN);
  k_enorm<<<nb(NE), T, 0, stream>>>(erow, ecol, dis, enorm, NE);

  static const int DIN[5]  = {39, 1024, 512, 256, 512};
  static const int DINP[5] = {64, 1024, 512, 256, 512};
  static const int DOUT[5] = {1024, 512, 256, 512, 1024};

  for (int l = 0; l < 5; ++l) {
    const float* src = (l == 0) ? x0 : bufX;
    int din = DIN[l], dinp = DINP[l], dout = DOUT[l];
    k_split<<<nb((long long)NN * dinp), T, 0, stream>>>(src, NN, din, dinp, Xhi, Xlo);
    k_wt<<<nb((long long)dout * dinp), T, 0, stream>>>(Wl[l], din, dinp, dout, WhiT, WloT);
    int tiles = (NN / 16) * (dout / 16);
    k_gemm<<<(tiles + 7) / 8, 256, 0, stream>>>(Xhi, Xlo, WhiT, WloT, bufH, NN, dinp, dout);
    k_selfinit<<<nb((long long)NN * dout), T, 0, stream>>>(bufH, dis, bl[l], bufX, NN, dout);
    k_scatter<<<nb((long long)NE * (dout / 4)), T, 0, stream>>>(erow, ecol, enorm, bufH, bufX, NE, dout);
    k_fill<<<nb(dout), T, 0, stream>>>((unsigned*)gsum, dout, 0u);
    k_fill<<<nb(dout), T, 0, stream>>>((unsigned*)gsqs, dout, 0u);
    k_bnstats<<<(NN + 255) / 256, 256, 0, stream>>>(bufX, NN, dout, gsum, gsqs);
    k_bnfinal<<<nb(dout), T, 0, stream>>>(gsum, gsqs, gl[l], bel[l], bscale, bshift, dout, NN);
    k_bnapply<<<nb((long long)NN * dout), T, 0, stream>>>(bufX, bscale, bshift, NN, dout);
  }

  // attention pooling over graphs
  auto fb = [&](long long waves) { return (unsigned)((waves + 7) / 8); };   // 8 waves/block
  k_fcw<<<fb((long long)NN), 256, 0, stream>>>(bufX, gateW, gateB, gate, NN, 1024, 1, 0);
  k_fill<<<nb(NG), T, 0, stream>>>(gmU, NG, 0x007FFFFFu);        // enc(-inf)
  k_fill<<<nb(NG), T, 0, stream>>>((unsigned*)denom, NG, 0u);
  k_gmax<<<nb(NN), T, 0, stream>>>(gate, batch, gmU, NN);
  k_exp<<<nb(NN), T, 0, stream>>>(gate, batch, gmU, eex, denom, NN);
  k_alpha<<<nb(NN), T, 0, stream>>>(eex, denom, batch, gate, NN);   // gate := alpha
  k_fill<<<nb((long long)NG * 1024), T, 0, stream>>>((unsigned*)pooled, (long long)NG * 1024, 0u);
  k_pool<<<nb((long long)NN * 256), T, 0, stream>>>(bufX, gate, batch, pooled, NN);

  // MLP head
  k_fcw<<<fb((long long)NG * 128), 256, 0, stream>>>(pooled, fc2W, fc2b, h2, NG, 1024, 128, 1);
  k_fcw<<<fb((long long)NG * 16), 256, 0, stream>>>(h2, fc3W, fc3b, h3, NG, 128, 16, 1);
  k_fcw<<<fb((long long)NG), 256, 0, stream>>>(h3, fc4W, fc4b, (float*)d_out, NG, 16, 1, 0);
}
